// ServerGPT2Model_36885179138284
// MI455X (gfx1250) — matrix operand order repaired and checked by reading, not compiled, not run on hardware
//
#include <hip/hip_runtime.h>

// ---------------------------------------------------------------------------
// GPT-2 forward (B=8, S=1024, D=768, H=12, HD=64, FF=3072, L=12) for gfx1250.
// bf16 data path + v_wmma_f32_16x16x32_bf16, fp32 accumulation, fused
// epilogues, async global->LDS (ASYNCcnt) double-buffered GEMM pipeline.
// ---------------------------------------------------------------------------

typedef __bf16 bf16;
typedef __bf16 v16bf __attribute__((ext_vector_type(16)));
typedef float  v8f   __attribute__((ext_vector_type(8)));

union Frag  { v16bf v; uint4 u[2]; };
union BVec8 { uint4 u; bf16 h[8]; };

__device__ __forceinline__ v8f wmma_bf16(const Frag& a, const Frag& b, v8f c) {
  return __builtin_amdgcn_wmma_f32_16x16x32_bf16(
      /*neg_a=*/false, a.v, /*neg_b=*/false, b.v,
      /*c_mod=*/(short)0, c, /*reuse_a=*/false, /*reuse_b=*/false);
}

// Async 16B global -> LDS copy (gfx1250, tracked with ASYNCcnt).
// LDS byte address = low 32 bits of the generic pointer (flat->LDS rule).
#define USE_ASYNC 1

__device__ __forceinline__ void async_ld16(const bf16* g, bf16* l) {
#if USE_ASYNC
  unsigned lds = (unsigned)(unsigned long long)l;
  asm volatile("global_load_async_to_lds_b128 %0, %1, off"
               :: "v"(lds), "v"(g) : "memory");
#else
  *(uint4*)l = *(const uint4*)g;
#endif
}
__device__ __forceinline__ void wait_async() {
#if USE_ASYNC
  asm volatile("s_wait_asynccnt 0x0" ::: "memory");
#endif
}

#define NTOK  8192          // B*S
#define DMODEL 768
#define NHEAD 12
#define HDIM  64
#define FFDIM 3072
#define QKVN  2304

// ---------------------------------------------------------------------------
// LayerNorm over D=768. One block (256 threads) per token.
// ---------------------------------------------------------------------------
__global__ __launch_bounds__(256) void layernorm_kernel(
    const float* __restrict__ x, const float* __restrict__ g,
    const float* __restrict__ be, bf16* __restrict__ outb,
    float* __restrict__ outf)
{
  const int row = blockIdx.x;
  const int tid = threadIdx.x;
  const float* xr = x + (size_t)row * DMODEL;
  __shared__ float red[256];

  float v0 = xr[tid], v1 = xr[tid + 256], v2 = xr[tid + 512];

  red[tid] = v0 + v1 + v2;
  __syncthreads();
  for (int o = 128; o > 0; o >>= 1) {
    if (tid < o) red[tid] += red[tid + o];
    __syncthreads();
  }
  float mu = red[0] * (1.0f / DMODEL);
  __syncthreads();

  float d0 = v0 - mu, d1 = v1 - mu, d2 = v2 - mu;
  red[tid] = d0 * d0 + d1 * d1 + d2 * d2;
  __syncthreads();
  for (int o = 128; o > 0; o >>= 1) {
    if (tid < o) red[tid] += red[tid + o];
    __syncthreads();
  }
  float rstd = rsqrtf(red[0] * (1.0f / DMODEL) + 1e-5f);

  float y0 = d0 * rstd * g[tid]       + be[tid];
  float y1 = d1 * rstd * g[tid + 256] + be[tid + 256];
  float y2 = d2 * rstd * g[tid + 512] + be[tid + 512];
  if (outb) {
    bf16* o = outb + (size_t)row * DMODEL;
    o[tid] = (bf16)y0; o[tid + 256] = (bf16)y1; o[tid + 512] = (bf16)y2;
  } else {
    float* o = outf + (size_t)row * DMODEL;
    o[tid] = y0; o[tid + 256] = y1; o[tid + 512] = y2;
  }
}

// ---------------------------------------------------------------------------
// Weight convert + transpose: W fp32 [K,N] -> Wt bf16 [N,K]
// ---------------------------------------------------------------------------
__global__ void convert_transpose_kernel(const float* __restrict__ W,
                                         bf16* __restrict__ Wt,
                                         int K, int N)
{
  size_t idx = (size_t)blockIdx.x * 256 + threadIdx.x;
  if (idx >= (size_t)K * N) return;
  int n = (int)(idx / K);
  int k = (int)(idx % K);
  Wt[idx] = (bf16)W[(size_t)k * N + n];
}

// ---------------------------------------------------------------------------
// Tiled bf16 GEMM, double-buffered LDS, async global->LDS prefetch.
//   C[M,N] = A[M,K](bf16) x Bt[N,K](bf16)^T  (fp32 accum)
// mode 0 (QKV):  out_bf16 = acc + bias[n] + lora[m,n]
// mode 1 (RES):  resid[m,n] += acc + bias[n]
// mode 2 (GELU): out_bf16 = gelu_tanh(acc + bias[n])
// Block tile 128x128x32; 8 waves in 2x4; per-wave 64x32 (4x2 wmma tiles).
// ---------------------------------------------------------------------------
#define BM 128
#define BN 128
#define BK 32
#define LDSPAD 8
#define BKP (BK + LDSPAD)

__global__ __launch_bounds__(256) void gemm_bf16_kernel(
    const bf16* __restrict__ A, const bf16* __restrict__ Bt,
    const float* __restrict__ bias, const float* __restrict__ lora,
    float* __restrict__ resid, bf16* __restrict__ outb,
    int M, int N, int K, int mode)
{
  __shared__ __align__(16) bf16 sA[2][BM][BKP];
  __shared__ __align__(16) bf16 sB[2][BN][BKP];

  const int tid  = threadIdx.x;
  const int wid  = tid >> 5;
  const int lane = tid & 31;
  const int l16  = lane & 15;
  const int hi   = lane >> 4;
  const int waveM = wid >> 2;   // 0..1
  const int waveN = wid & 3;    // 0..3
  const int bm = blockIdx.y * BM;
  const int bn = blockIdx.x * BN;

  // per-thread copy slots: 2 chunks of 16B per tile per matrix
  const int rT = tid >> 2;            // 0..63
  const int cT = (tid & 3) * 8;       // 0,8,16,24
  const bf16* gA0 = A  + (size_t)(bm + rT) * K + cT;
  const bf16* gA1 = gA0 + (size_t)64 * K;
  const bf16* gB0 = Bt + (size_t)(bn + rT) * K + cT;
  const bf16* gB1 = gB0 + (size_t)64 * K;

  v8f acc[4][2];
  for (int i = 0; i < 4; ++i)
    for (int j = 0; j < 2; ++j)
      for (int e = 0; e < 8; ++e) acc[i][j][e] = 0.0f;

  const int nk = K / BK;
  // prefetch tile 0 into buffer 0
  {
    async_ld16(gA0, &sA[0][rT][cT]);
    async_ld16(gA1, &sA[0][rT + 64][cT]);
    async_ld16(gB0, &sB[0][rT][cT]);
    async_ld16(gB1, &sB[0][rT + 64][cT]);
  }

  for (int kt = 0; kt < nk; ++kt) {
    const int buf = kt & 1;
    wait_async();
    __syncthreads();
    if (kt + 1 < nk) {   // prefetch next tile into other buffer
      const size_t ko = (size_t)(kt + 1) * BK;
      async_ld16(gA0 + ko, &sA[buf ^ 1][rT][cT]);
      async_ld16(gA1 + ko, &sA[buf ^ 1][rT + 64][cT]);
      async_ld16(gB0 + ko, &sB[buf ^ 1][rT][cT]);
      async_ld16(gB1 + ko, &sB[buf ^ 1][rT + 64][cT]);
    }

    // fragments per ISA lane layout
    Frag af[4], bfr[2];
#pragma unroll
    for (int i = 0; i < 4; ++i) {
      int r = waveM * 64 + i * 16 + l16;
      af[i].u[0] = *(const uint4*)&sA[buf][r][8 * hi];
      af[i].u[1] = *(const uint4*)&sA[buf][r][16 + 8 * hi];
    }
#pragma unroll
    for (int j = 0; j < 2; ++j) {
      int r = waveN * 32 + j * 16 + l16;
      bfr[j].u[0] = *(const uint4*)&sB[buf][r][8 * hi];
      bfr[j].u[1] = *(const uint4*)&sB[buf][r][16 + 8 * hi];
    }
#pragma unroll
    for (int i = 0; i < 4; ++i)
#pragma unroll
      for (int j = 0; j < 2; ++j)
        acc[i][j] = wmma_bf16(af[i], bfr[j], acc[i][j]);
  }

  // epilogue (C layout: M = r + 8*hi, N = lane&15)
  for (int i = 0; i < 4; ++i)
    for (int j = 0; j < 2; ++j) {
      int col = bn + waveN * 32 + j * 16 + l16;
      float bv = bias[col];
      for (int r = 0; r < 8; ++r) {
        int row = bm + waveM * 64 + i * 16 + hi * 8 + r;
        float v = acc[i][j][r] + bv;
        if (mode == 0) {
          v += lora[(size_t)row * N + col];
          outb[(size_t)row * N + col] = (bf16)v;
        } else if (mode == 1) {
          resid[(size_t)row * N + col] += v;
        } else {
          float t  = v + 0.044715f * v * v * v;
          float gl = 0.5f * v * (1.0f + tanhf(0.7978845608028654f * t));
          outb[(size_t)row * N + col] = (bf16)gl;
        }
      }
    }
}

// ---------------------------------------------------------------------------
// Flash attention: one block (8 waves) per (b, h, 128-query tile).
// Wave w owns q rows [16w,16w+16): per-wave online softmax in registers.
// K tile copied with async global->LDS (pure 16B-chunk copy); V transposed
// through registers.
// ---------------------------------------------------------------------------
__global__ __launch_bounds__(256) void attn_kernel(
    const bf16* __restrict__ qkv, bf16* __restrict__ o)
{
  const int qt = blockIdx.x;   // 0..7 (query tile of 128)
  const int h  = blockIdx.y;   // 0..11
  const int b  = blockIdx.z;   // 0..7
  const int tid = threadIdx.x, wid = tid >> 5, lane = tid & 31;
  const int l16 = lane & 15, hi = lane >> 4;
  const size_t RS = QKVN;      // qkv row stride (elements)

  __shared__ __align__(16) bf16 sK [64][72];      // [key][d]
  __shared__ __align__(16) bf16 sVt[64][72];      // [d][key]
  __shared__ __align__(16) bf16 sP [8][16][72];   // per-wave P handoff

  // Q fragments (A-layout) straight from global: lane = q row in wave tile
  const int qrow0 = qt * 128 + wid * 16;
  const size_t mQ = (size_t)(b * 1024 + qrow0 + l16) * RS + h * HDIM;
  Frag qA[2];
  qA[0].u[0] = *(const uint4*)&qkv[mQ + 8 * hi];
  qA[0].u[1] = *(const uint4*)&qkv[mQ + 16 + 8 * hi];
  qA[1].u[0] = *(const uint4*)&qkv[mQ + 32 + 8 * hi];
  qA[1].u[1] = *(const uint4*)&qkv[mQ + 32 + 16 + 8 * hi];

  v8f Oacc[4];
  for (int d = 0; d < 4; ++d)
    for (int e = 0; e < 8; ++e) Oacc[d][e] = 0.0f;
  float mrow[8], lrow[8];
  for (int r = 0; r < 8; ++r) { mrow[r] = -1e30f; lrow[r] = 0.0f; }

  // per-thread K/V copy slots (2 x 16B chunks each)
  const int krT = tid >> 3;           // 0..31
  const int cKT = (tid & 7) * 8;      // 0..56
  const bf16* gKbase = qkv + (size_t)(b * 1024 + krT) * RS + DMODEL + h * HDIM + cKT;
  const bf16* gVbase = qkv + (size_t)(b * 1024 + krT) * RS + 2 * DMODEL + h * HDIM + cKT;

  const int nkb = 2 * qt + 2;              // causal pruning (64-key blocks)
  for (int kb = 0; kb < nkb; ++kb) {
    __syncthreads();
    // async copy K block (natural [key][d] layout == transposed-B layout)
    {
      const bf16* gK = gKbase + (size_t)(kb * 64) * RS;
      async_ld16(gK,                    &sK[krT][cKT]);
      async_ld16(gK + (size_t)32 * RS,  &sK[krT + 32][cKT]);
    }
    // V block transposed -> [d][key] (register staging)
    {
      const bf16* gV = gVbase + (size_t)(kb * 64) * RS;
      BVec8 v0, v1;
      v0.u = *(const uint4*)gV;
      v1.u = *(const uint4*)(gV + (size_t)32 * RS);
#pragma unroll
      for (int e = 0; e < 8; ++e) {
        sVt[cKT + e][krT]      = v0.h[e];
        sVt[cKT + e][krT + 32] = v1.h[e];
      }
    }
    wait_async();
    __syncthreads();

    // S = Q K^T   (4 key subtiles x 2 d-chunks)
    v8f sAcc[4];
#pragma unroll
    for (int j = 0; j < 4; ++j) {
      for (int e = 0; e < 8; ++e) sAcc[j][e] = 0.0f;
#pragma unroll
      for (int c = 0; c < 2; ++c) {
        Frag bK;
        bK.u[0] = *(const uint4*)&sK[j * 16 + l16][c * 32 + 8 * hi];
        bK.u[1] = *(const uint4*)&sK[j * 16 + l16][c * 32 + 16 + 8 * hi];
        sAcc[j] = wmma_bf16(qA[c], bK, sAcc[j]);
      }
    }

    // scale + causal mask + per-row max
    float pmax[8];
    for (int r = 0; r < 8; ++r) pmax[r] = -1e30f;
    for (int j = 0; j < 4; ++j) {
      int key = kb * 64 + j * 16 + l16;
      for (int r = 0; r < 8; ++r) {
        int q = qrow0 + hi * 8 + r;
        float s = sAcc[j][r] * 0.125f;        // 1/sqrt(64)
        s = (key <= q) ? s : -1e30f;
        sAcc[j][r] = s;
        pmax[r] = fmaxf(pmax[r], s);
      }
    }
    for (int msk = 1; msk <= 8; msk <<= 1)
      for (int r = 0; r < 8; ++r)
        pmax[r] = fmaxf(pmax[r], __shfl_xor(pmax[r], msk, 32));

    // online softmax update
    float alpha[8], rsum[8];
    for (int r = 0; r < 8; ++r) {
      float mnew = fmaxf(mrow[r], pmax[r]);
      alpha[r] = __expf(mrow[r] - mnew);
      mrow[r] = mnew;
      rsum[r] = 0.0f;
    }
    for (int j = 0; j < 4; ++j)
      for (int r = 0; r < 8; ++r) {
        float p = __expf(sAcc[j][r] - mrow[r]);
        sAcc[j][r] = p;
        rsum[r] += p;
      }
    for (int msk = 1; msk <= 8; msk <<= 1)
      for (int r = 0; r < 8; ++r)
        rsum[r] += __shfl_xor(rsum[r], msk, 32);
    for (int r = 0; r < 8; ++r) lrow[r] = lrow[r] * alpha[r] + rsum[r];
    for (int d = 0; d < 4; ++d)
      for (int r = 0; r < 8; ++r) Oacc[d][r] *= alpha[r];

    // hand P (C-layout) to LDS as A-layout source
    for (int j = 0; j < 4; ++j)
      for (int r = 0; r < 8; ++r)
        sP[wid][hi * 8 + r][j * 16 + l16] = (bf16)sAcc[j][r];
    __syncthreads();

    // O += P V
#pragma unroll
    for (int kc = 0; kc < 2; ++kc) {
      Frag aP;
      aP.u[0] = *(const uint4*)&sP[wid][l16][kc * 32 + 8 * hi];
      aP.u[1] = *(const uint4*)&sP[wid][l16][kc * 32 + 16 + 8 * hi];
#pragma unroll
      for (int db = 0; db < 4; ++db) {
        Frag bV;
        bV.u[0] = *(const uint4*)&sVt[db * 16 + l16][kc * 32 + 8 * hi];
        bV.u[1] = *(const uint4*)&sVt[db * 16 + l16][kc * 32 + 16 + 8 * hi];
        Oacc[db] = wmma_bf16(aP, bV, Oacc[db]);
      }
    }
  }

  // normalize + store (merge heads: o[b, s, h*64 + d])
  for (int db = 0; db < 4; ++db)
    for (int r = 0; r < 8; ++r) {
      int srow = qrow0 + hi * 8 + r;
      size_t oi = (size_t)(b * 1024 + srow) * DMODEL + h * HDIM + db * 16 + l16;
      o[oi] = (bf16)(Oacc[db][r] / lrow[r]);
    }
}

// ---------------------------------------------------------------------------
// Host orchestration
// ---------------------------------------------------------------------------
extern "C" void kernel_launch(void* const* d_in, const int* in_sizes, int n_in,
                              void* d_out, int out_size, void* d_ws, size_t ws_size,
                              hipStream_t stream) {
  (void)in_sizes; (void)n_in; (void)out_size; (void)ws_size;
  const float* hid     = (const float*)d_in[0];
  const float* lora    = (const float*)d_in[1];
  const float* ln1_g   = (const float*)d_in[2];
  const float* ln1_b   = (const float*)d_in[3];
  const float* W_attn  = (const float*)d_in[4];
  const float* b_attn  = (const float*)d_in[5];
  const float* W_aproj = (const float*)d_in[6];
  const float* b_aproj = (const float*)d_in[7];
  const float* ln2_g   = (const float*)d_in[8];
  const float* ln2_b   = (const float*)d_in[9];
  const float* W_fc    = (const float*)d_in[10];
  const float* b_fc    = (const float*)d_in[11];
  const float* W_mproj = (const float*)d_in[12];
  const float* b_mproj = (const float*)d_in[13];
  const float* lnf_g   = (const float*)d_in[14];
  const float* lnf_b   = (const float*)d_in[15];

  // workspace layout (256B aligned)
  char* ws = (char*)d_ws;
  size_t off = 0;
  auto take = [&](size_t bytes) {
    size_t o = off;
    off = (off + bytes + 255) & ~(size_t)255;
    return o;
  };
  float* X    = (float*)(ws + take((size_t)NTOK * DMODEL * 4));
  bf16*  LN   = (bf16*) (ws + take((size_t)NTOK * DMODEL * 2));
  bf16*  QKV  = (bf16*) (ws + take((size_t)NTOK * QKVN * 2));
  bf16*  O    = (bf16*) (ws + take((size_t)NTOK * DMODEL * 2));
  bf16*  MACT = (bf16*) (ws + take((size_t)NTOK * FFDIM * 2));
  bf16*  WT0  = (bf16*) (ws + take((size_t)DMODEL * QKVN * 2));   // [2304][768]
  bf16*  WT1  = (bf16*) (ws + take((size_t)DMODEL * DMODEL * 2)); // [768][768]
  bf16*  WT2  = (bf16*) (ws + take((size_t)DMODEL * FFDIM * 2));  // [3072][768]
  bf16*  WT3  = (bf16*) (ws + take((size_t)FFDIM * DMODEL * 2));  // [768][3072]

  // x = hidden_states (don't mutate inputs)
  hipMemcpyAsync(X, hid, (size_t)NTOK * DMODEL * 4,
                 hipMemcpyDeviceToDevice, stream);

  auto cvt_blocks = [](size_t n) { return (unsigned)((n + 255) / 256); };

  for (int l = 0; l < 12; ++l) {
    const float* Wa = W_attn  + (size_t)l * DMODEL * QKVN;
    const float* Wp = W_aproj + (size_t)l * DMODEL * DMODEL;
    const float* Wf = W_fc    + (size_t)l * DMODEL * FFDIM;
    const float* Wm = W_mproj + (size_t)l * FFDIM * DMODEL;

    // --- attention ---
    layernorm_kernel<<<NTOK, 256, 0, stream>>>(
        X, ln1_g + l * DMODEL, ln1_b + l * DMODEL, LN, nullptr);

    convert_transpose_kernel<<<cvt_blocks((size_t)DMODEL * QKVN), 256, 0, stream>>>(
        Wa, WT0, DMODEL, QKVN);
    gemm_bf16_kernel<<<dim3(QKVN / BN, NTOK / BM), 256, 0, stream>>>(
        LN, WT0, b_attn + (size_t)l * QKVN, lora, nullptr, QKV,
        NTOK, QKVN, DMODEL, /*mode=*/0);

    attn_kernel<<<dim3(8, NHEAD, 8), 256, 0, stream>>>(QKV, O);

    convert_transpose_kernel<<<cvt_blocks((size_t)DMODEL * DMODEL), 256, 0, stream>>>(
        Wp, WT1, DMODEL, DMODEL);
    gemm_bf16_kernel<<<dim3(DMODEL / BN, NTOK / BM), 256, 0, stream>>>(
        O, WT1, b_aproj + (size_t)l * DMODEL, nullptr, X, nullptr,
        NTOK, DMODEL, DMODEL, /*mode=*/1);

    // --- MLP ---
    layernorm_kernel<<<NTOK, 256, 0, stream>>>(
        X, ln2_g + l * DMODEL, ln2_b + l * DMODEL, LN, nullptr);

    convert_transpose_kernel<<<cvt_blocks((size_t)DMODEL * FFDIM), 256, 0, stream>>>(
        Wf, WT2, DMODEL, FFDIM);
    gemm_bf16_kernel<<<dim3(FFDIM / BN, NTOK / BM), 256, 0, stream>>>(
        LN, WT2, b_fc + (size_t)l * FFDIM, nullptr, nullptr, MACT,
        NTOK, FFDIM, DMODEL, /*mode=*/2);

    convert_transpose_kernel<<<cvt_blocks((size_t)FFDIM * DMODEL), 256, 0, stream>>>(
        Wm, WT3, FFDIM, DMODEL);
    gemm_bf16_kernel<<<dim3(DMODEL / BN, NTOK / BM), 256, 0, stream>>>(
        MACT, WT3, b_mproj + (size_t)l * DMODEL, nullptr, X, nullptr,
        NTOK, DMODEL, FFDIM, /*mode=*/1);
  }

  // final LayerNorm -> fp32 output
  layernorm_kernel<<<NTOK, 256, 0, stream>>>(
      X, lnf_g, lnf_b, nullptr, (float*)d_out);
}
